// DCA_57226144252734
// MI455X (gfx1250) — compile-verified, matrix operand order checked by reading, not run on hardware
//
#include <hip/hip_runtime.h>
#include <hip/hip_bf16.h>

// ===========================================================================
// MI455X / gfx1250 dual-attention block (channel XCA + Linformer spatial).
//
// Roofline: ~50 GFLOP vs ~300 MB traffic -> matrix-core bound. All
// contractions use v_wmma_f32_16x16x32_f16 (wave32, f32 accum); softmax /
// L2-norm math stays f32. Intermediates stored f16 in workspace (~304 MB).
// Final projections stage their (already-f16) A tiles into LDS with the
// Tensor Data Mover (tensor_load_to_lds + s_wait_tensorcnt, 6-arg builtin).
//
// B=16, N=4096 tokens, C=256, H=4 heads, HD=64, P=64.
// ===========================================================================

#define NB   16
#define NTOK 4096
#define NC   256
#define NH   4
#define HD   64
#define NP   64

typedef __attribute__((ext_vector_type(16))) _Float16 v16h;
typedef __attribute__((ext_vector_type(8)))  float    v8f;
typedef __attribute__((ext_vector_type(8)))  unsigned v8u;
typedef __attribute__((ext_vector_type(4)))  unsigned v4u;
typedef __attribute__((ext_vector_type(8)))  int      v8i;
typedef __attribute__((ext_vector_type(4)))  int      v4i;

#if defined(__has_builtin)
#if __has_builtin(__builtin_amdgcn_tensor_load_to_lds) && \
    __has_builtin(__builtin_amdgcn_s_wait_tensorcnt)
#define USE_TDM 1
#endif
#endif
#ifndef USE_TDM
#define USE_TDM 0
#endif

static __device__ __forceinline__ v8f wmma_f16(v16h a, v16h b, v8f c) {
  // D = A(16x32 f16) x B(32x16 f16) + C(16x16 f32)
  return __builtin_amdgcn_wmma_f32_16x16x32_f16(
      /*neg_a=*/false, a, /*neg_b=*/false, b,
      /*c_mod=*/(short)0, c, /*reuse_a=*/false, /*reuse_b=*/false);
}

static __device__ __forceinline__ v8f zero8() {
  v8f z = {0.f, 0.f, 0.f, 0.f, 0.f, 0.f, 0.f, 0.f};
  return z;
}

// Build one 16x32 f16 fragment. Lane-dim (lane&15) indexes 16 stored rows
// (stride ld elements), K-dim indexes 32 contiguous columns. Matches the ISA
// 16-bit A/B layout: lane L holds row L%16; VGPR v holds K pair
// (v/4)*16 + (L/16)*8 + (v%4)*2. Pairs are contiguous -> dword loads.
// Works for global or LDS pointers (inlined, addrspace inferred).
static __device__ __forceinline__ v16h load_frag(const _Float16* base, int ld) {
  const int lane = threadIdx.x & 31;
  const int r    = lane & 15;
  const int hf   = lane >> 4;
  const _Float16* row = base + (size_t)r * ld;
  v8u u;
#pragma unroll
  for (int v = 0; v < 8; ++v) {
    const int k = ((v >> 2) << 4) + (hf << 3) + ((v & 3) << 1);
    u[v] = *(const unsigned*)(row + k);
  }
  return __builtin_bit_cast(v16h, u);
}

// ---------------------------------------------------------------------------
// f32 -> f16 elementwise convert (for WE).
// ---------------------------------------------------------------------------
__global__ __launch_bounds__(256) void cvt_f32_f16_kernel(
    const float* __restrict__ src, _Float16* __restrict__ dst, int n) {
  int i = blockIdx.x * 256 + threadIdx.x;
  if (i < n) dst[i] = (_Float16)src[i];
}

// ---------------------------------------------------------------------------
// Tiled WMMA GEMM:  out = A(M x K) * W(Nc x K)^T + bias
// Block tile 128(M) x 128(N) x 32(K), 8 waves (2x4), each wave 64x32 = 8 WMMA
// accumulators. A/W staged to LDS as f16. 256 threads.
//   MODE 0: out_f[m*256 + nc]                     (final projections, f32)
//   MODE 1: q  -> channel-major f16 [b,h,hd,n]
//   MODE 2: kvv-> t=0 (k, ch-major) | t=1 (v_ca, token-major) | t=2 (v_sa)
// For AF16 (A already f16 in memory) the A tile is moved by the Tensor Data
// Mover: a 2-D D# (tile 32x128, data_size=2B) with LDS padding
// (pad_interval=16 DW, pad_amount=2 DW) reproducing the As[128][36] layout.
// ---------------------------------------------------------------------------
template <int MODE, bool AF16>
__global__ __launch_bounds__(256) void gemm_kernel(
    const void* __restrict__ Ap, const float* __restrict__ Wg,
    const float* __restrict__ bias, int Kdim,
    float* __restrict__ outf, _Float16* __restrict__ outh) {
  __shared__ _Float16 As[128][36];
  __shared__ _Float16 Bs[128][36];
  const int tid  = threadIdx.x;
  const int m0   = blockIdx.x * 128;
  const int n0   = blockIdx.y * 128;
  const int w    = tid >> 5, lane = tid & 31;
  const int r    = lane & 15, hf = lane >> 4;
  const int wm   = w & 1, wn = w >> 1;

  v8f acc[4][2];
#pragma unroll
  for (int mi = 0; mi < 4; ++mi)
#pragma unroll
    for (int ni = 0; ni < 2; ++ni) acc[mi][ni] = zero8();

  for (int k0 = 0; k0 < Kdim; k0 += 32) {
    if constexpr (AF16) {
#if USE_TDM
      // ---- TDM: async A-tile (f16) global -> LDS, issued by wave 0 only.
      if (w == 0) {
        const unsigned long long ga =
            (unsigned long long)Ap + ((size_t)m0 * Kdim + k0) * 2ull;
        const unsigned ldsa = (unsigned)(unsigned long long)(void*)&As[0][0];
        v4u g0;
        g0[0] = 1u;  // count=1 (valid user descriptor), gather off
        g0[1] = ldsa;                                   // lds_addr
        g0[2] = (unsigned)ga;                           // global_addr[31:0]
        g0[3] = (unsigned)((ga >> 32) & 0x01FFFFFFull)  // global_addr[56:32]
                | (2u << 30);                           // type=2 (image)
        v8i g1;
        // data_size=2B (1<<16), pad_enable (1<<20), pad_interval=16DW (3<<22),
        // pad_amount=2DW (1<<25); workgroup_mask=0 (not in cluster).
        g1[0] = (int)((1u << 16) | (1u << 20) | (3u << 22) | (1u << 25));
        g1[1] = (int)(((unsigned)Kdim & 0xFFFFu) << 16);       // tensor_dim0 lo
        g1[2] = (int)((((unsigned)Kdim >> 16) & 0xFFFFu));     // dim0 hi, dim1 lo=0
        g1[3] = (int)((65536u >> 16) | (32u << 16));           // dim1 hi | tile_dim0=32
        g1[4] = 128;                                           // tile_dim1=128
        g1[5] = Kdim;                                          // dim0_stride lo
        g1[6] = 0;                                             // stride hi
        g1[7] = 0;
        v4i g2 = {0, 0, 0, 0};  // tile_dim2..4 = 0 -> 2-D transfer
        v4i g3 = {0, 0, 0, 0};
        v8i g4 = {0, 0, 0, 0, 0, 0, 0, 0};
        __builtin_amdgcn_tensor_load_to_lds(g0, g1, g2, g3, g4, 0);
      }
#else
#pragma unroll
      for (int i = 0; i < 16; ++i) {
        const int idx = tid + (i << 8);
        const int rr = idx >> 5, cc = idx & 31;
        As[rr][cc] = ((const _Float16*)Ap)[(size_t)(m0 + rr) * Kdim + k0 + cc];
      }
#endif
      // Weight tile still needs f32->f16 conversion on the VALU.
#pragma unroll
      for (int i = 0; i < 16; ++i) {
        const int idx = tid + (i << 8);
        const int rr = idx >> 5, cc = idx & 31;
        Bs[rr][cc] = (_Float16)Wg[(size_t)(n0 + rr) * Kdim + k0 + cc];
      }
#if USE_TDM
      if (w == 0) __builtin_amdgcn_s_wait_tensorcnt(0);
#endif
    } else {
#pragma unroll
      for (int i = 0; i < 16; ++i) {
        const int idx = tid + (i << 8);
        const int rr = idx >> 5, cc = idx & 31;
        As[rr][cc] =
            (_Float16)((const float*)Ap)[(size_t)(m0 + rr) * Kdim + k0 + cc];
        Bs[rr][cc] = (_Float16)Wg[(size_t)(n0 + rr) * Kdim + k0 + cc];
      }
    }
    if (k0 + 32 < Kdim) {  // global_prefetch_b8 of the next weight K-slab
      __builtin_prefetch(
          (const char*)(Wg + (size_t)(n0 + (tid & 127)) * Kdim + k0 + 32), 0, 0);
    }
    __syncthreads();

    v16h bfr0 = load_frag(&Bs[wn * 32][0], 36);
    v16h bfr1 = load_frag(&Bs[wn * 32 + 16][0], 36);
#pragma unroll
    for (int mi = 0; mi < 4; ++mi) {
      v16h af = load_frag(&As[wm * 64 + mi * 16][0], 36);
      acc[mi][0] = wmma_f16(af, bfr0, acc[mi][0]);
      acc[mi][1] = wmma_f16(af, bfr1, acc[mi][1]);
    }
    __syncthreads();
  }

  const size_t TS = (size_t)NB * NH * HD * NTOK;  // elems per qkv tensor
#pragma unroll
  for (int mi = 0; mi < 4; ++mi)
#pragma unroll
    for (int ni = 0; ni < 2; ++ni)
#pragma unroll
      for (int i = 0; i < 8; ++i) {
        const int m  = m0 + wm * 64 + mi * 16 + i + (hf << 3);
        const int nc = n0 + wn * 32 + ni * 16 + r;
        const float val = acc[mi][ni][i] + bias[nc];
        if constexpr (MODE == 0) {
          outf[(size_t)m * NC + nc] = val;
        } else if constexpr (MODE == 1) {
          const int b = m >> 12, n = m & (NTOK - 1);
          const int hh = nc >> 6, hd = nc & 63;
          outh[(((size_t)(b * NH + hh)) * HD + hd) * NTOK + n] = (_Float16)val;
        } else {
          const int b = m >> 12, n = m & (NTOK - 1);
          const int t = nc >> 8, c = nc & 255;
          const int hh = c >> 6, hd = c & 63;
          const size_t bh = (size_t)(b * NH + hh);
          size_t idx2;
          if (t == 1)  // v_ca: token-major [bh][n][hd]
            idx2 = TS + ((bh << 12) + n) * HD + hd;
          else         // k / v_sa: channel-major [bh][hd][n]
            idx2 = (size_t)t * TS + ((bh << 6) + hd) * NTOK + n;
          outh[idx2] = (_Float16)val;
        }
      }
}

// ---------------------------------------------------------------------------
// L2 norm over tokens + normalize. One workgroup per (b,h,hd) row.
// blockIdx.y: 0 = q (emit ch-major + token-major), 1 = k (ch-major).
// ---------------------------------------------------------------------------
__global__ __launch_bounds__(256) void norm_kernel(
    const _Float16* __restrict__ qraw, const _Float16* __restrict__ kraw,
    _Float16* __restrict__ qn, _Float16* __restrict__ kn,
    _Float16* __restrict__ qnt) {
  __shared__ float red[256];
  const int tid = threadIdx.x;
  const int row = blockIdx.x;   // (b*H+h)*64 + hd
  const int sel = blockIdx.y;
  const _Float16* p = (sel ? kraw : qraw) + (size_t)row * NTOK;

  float s = 0.f;
  for (int n = tid; n < NTOK; n += 256) {
    float v = (float)p[n];
    s += v * v;
  }
  red[tid] = s;
  __syncthreads();
  for (int st = 128; st > 0; st >>= 1) {
    if (tid < st) red[tid] += red[tid + st];
    __syncthreads();
  }
  const float inv = 1.f / fmaxf(sqrtf(red[0]), 1e-12f);
  const int bh = row >> 6, hd = row & 63;
  for (int n = tid; n < NTOK; n += 256) {
    _Float16 v = (_Float16)((float)p[n] * inv);
    if (sel == 0) {
      qn[(size_t)row * NTOK + n] = v;
      qnt[((size_t)bh * NTOK + n) * HD + hd] = v;   // token-major
    } else {
      kn[(size_t)row * NTOK + n] = v;
    }
  }
}

// ---------------------------------------------------------------------------
// Channel attention per (b,h):
//   S = qn * kn^T (64x64, K=4096) -> softmax rows -> x_ca = P * v_ca (64xN).
// qn/kn channel-major; v_ca token-major so B-fragment K pairs are contiguous.
// ---------------------------------------------------------------------------
__global__ __launch_bounds__(256) void ca_attn_kernel(
    const _Float16* __restrict__ qn, const _Float16* __restrict__ kn,
    const _Float16* __restrict__ vca_t, const float* __restrict__ temp,
    _Float16* __restrict__ xca) {
  __shared__ float    Sm[64][65];
  __shared__ _Float16 Pm[64][66];
  const int bh = blockIdx.x, b = bh >> 2, h = bh & 3;
  const int tid = threadIdx.x, w = tid >> 5, lane = tid & 31;
  const int r = lane & 15, hf = lane >> 4;
  const size_t cbase = (size_t)bh * HD * NTOK;

  // Phase A: S = qn * kn^T, 16 tiles over 8 waves (2 each), K loop 4096.
  for (int t = w; t < 16; t += 8) {
    const int mi = t >> 2, ni = t & 3;
    v8f acc = zero8();
    for (int k0 = 0; k0 < NTOK; k0 += 32) {
      v16h af = load_frag(qn + cbase + (size_t)(mi * 16) * NTOK + k0, NTOK);
      v16h bf = load_frag(kn + cbase + (size_t)(ni * 16) * NTOK + k0, NTOK);
      acc = wmma_f16(af, bf, acc);
    }
    const float tv = temp[h];
#pragma unroll
    for (int i = 0; i < 8; ++i)
      Sm[mi * 16 + i + (hf << 3)][ni * 16 + r] = acc[i] * tv;
  }
  __syncthreads();

  // Phase B: row softmax (64 rows, one thread each).
  if (tid < 64) {
    float mx = Sm[tid][0];
    for (int c = 1; c < 64; ++c) mx = fmaxf(mx, Sm[tid][c]);
    float s = 0.f;
    for (int c = 0; c < 64; ++c) s += __expf(Sm[tid][c] - mx);
    const float inv = 1.f / s;
    for (int c = 0; c < 64; ++c)
      Pm[tid][c] = (_Float16)(__expf(Sm[tid][c] - mx) * inv);
  }
  __syncthreads();

  // Phase C: x_ca = P(64x64) * v_ca(64xN); A-fragments from LDS (ds loads).
  v16h ap[4][2];
#pragma unroll
  for (int mi = 0; mi < 4; ++mi) {
    ap[mi][0] = load_frag(&Pm[mi * 16][0], 66);
    ap[mi][1] = load_frag(&Pm[mi * 16][32], 66);
  }
  const size_t vbase = (size_t)bh * NTOK * HD;  // [n][hd]
  for (int ct = w; ct < NTOK / 16; ct += 8) {
    v16h bf0 = load_frag(vca_t + vbase + (size_t)(ct * 16) * HD, HD);
    v16h bf1 = load_frag(vca_t + vbase + (size_t)(ct * 16) * HD + 32, HD);
#pragma unroll
    for (int mi = 0; mi < 4; ++mi) {
      v8f acc = zero8();
      acc = wmma_f16(ap[mi][0], bf0, acc);
      acc = wmma_f16(ap[mi][1], bf1, acc);
#pragma unroll
      for (int i = 0; i < 8; ++i) {
        const int n = ct * 16 + r;                    // token (C col)
        const int c = mi * 16 + i + (hf << 3);        // head channel (C row)
        xca[((size_t)(b << 12) + n) * NC + h * HD + c] = (_Float16)acc[i];
      }
    }
  }
}

// ---------------------------------------------------------------------------
// Linformer projections per (b,h):
//   k_proj[c][p] = sum_n kn[c,n]*WE[p,n] + bE[p]  (stored [p][c] f16)
//   v_proj[c][p] = sum_n vsa[c,n]*WE[p,n] + bE[p] (stored [c][p] f16)
// 32 output tiles (2 tensors x 16) over 8 waves, K=4096.
// ---------------------------------------------------------------------------
__global__ __launch_bounds__(256) void linproj_kernel(
    const _Float16* __restrict__ kn, const _Float16* __restrict__ vsa,
    const _Float16* __restrict__ WEh, const float* __restrict__ bE,
    _Float16* __restrict__ kprojT, _Float16* __restrict__ vproj) {
  const int bh = blockIdx.x;
  const int tid = threadIdx.x, w = tid >> 5, lane = tid & 31;
  const int r = lane & 15, hf = lane >> 4;
  for (int tt = w * 4; tt < w * 4 + 4; ++tt) {
    const int which = tt >> 4, t = tt & 15, mi = t >> 2, ni = t & 3;
    const _Float16* src = which ? vsa : kn;
    v8f acc = zero8();
    const size_t abase = ((size_t)bh * HD + mi * 16) * NTOK;
    for (int k0 = 0; k0 < NTOK; k0 += 32) {
      v16h af = load_frag(src + abase + k0, NTOK);
      v16h bf = load_frag(WEh + (size_t)(ni * 16) * NTOK + k0, NTOK);
      acc = wmma_f16(af, bf, acc);
    }
#pragma unroll
    for (int i = 0; i < 8; ++i) {
      const int c = mi * 16 + i + (hf << 3);
      const int p = ni * 16 + r;
      const float val = acc[i] + bE[p];
      if (which == 0) kprojT[(size_t)bh * 4096 + p * HD + c] = (_Float16)val;
      else            vproj [(size_t)bh * 4096 + c * NP + p] = (_Float16)val;
    }
  }
}

// ---------------------------------------------------------------------------
// Spatial attention per (b,h):
//   S2 = qn^T * k_proj (NxP), softmax rows, x_sa = P2 * v_proj^T (NxHD),
//   then the reference's faithful scrambled permute on store.
// k_proj/v_proj live in LDS; each wave processes 16-token tiles.
// ---------------------------------------------------------------------------
__global__ __launch_bounds__(256) void sa_attn_kernel(
    const _Float16* __restrict__ qnt, const _Float16* __restrict__ kprojT,
    const _Float16* __restrict__ vproj, const float* __restrict__ temp2,
    _Float16* __restrict__ xsa) {
  __shared__ _Float16 kT[64][66];      // [p][c]
  __shared__ _Float16 vp[64][66];      // [c][p]
  __shared__ float    Srow[8][16][68]; // per-wave score tile (f16 view reused)
  const int bh = blockIdx.x, b = bh >> 2, h = bh & 3;
  const int tid = threadIdx.x, w = tid >> 5, lane = tid & 31;
  const int r = lane & 15, hf = lane >> 4;

  for (int i = tid; i < 4096; i += 256) {
    const int rr = i >> 6, cc = i & 63;
    kT[rr][cc] = kprojT[(size_t)bh * 4096 + i];
    vp[rr][cc] = vproj[(size_t)bh * 4096 + i];
  }
  __syncthreads();

  const float t2 = temp2[h];
  const size_t qbase = (size_t)bh * NTOK * HD;  // token-major [n][c]
  for (int j = 0; j < 32; ++j) {
    const int m0 = (w + 8 * j) * 16;  // 16-token row tile
    v16h aq0 = load_frag(qnt + qbase + (size_t)m0 * HD, HD);
    v16h aq1 = load_frag(qnt + qbase + (size_t)m0 * HD + 32, HD);
#pragma unroll
    for (int ct = 0; ct < 4; ++ct) {
      v8f acc = zero8();
      acc = wmma_f16(aq0, load_frag(&kT[ct * 16][0], 66), acc);
      acc = wmma_f16(aq1, load_frag(&kT[ct * 16][32], 66), acc);
#pragma unroll
      for (int i = 0; i < 8; ++i)
        Srow[w][i + (hf << 3)][ct * 16 + r] = acc[i] * t2;
    }
    __syncthreads();

    if (lane < 16) {  // one row of 64 per lane; write f16 P in-place
      float* row = &Srow[w][lane][0];
      float mx = row[0];
      for (int c = 1; c < 64; ++c) mx = fmaxf(mx, row[c]);
      float s = 0.f;
      for (int c = 0; c < 64; ++c) s += __expf(row[c] - mx);
      const float inv = 1.f / s;
      _Float16* prow = (_Float16*)row;  // f16 view; c/2 <= c so safe ascending
      for (int c = 0; c < 64; ++c) {
        const float e = __expf(row[c] - mx) * inv;
        prow[c] = (_Float16)e;
      }
    }
    __syncthreads();

    const _Float16* pbase = (const _Float16*)&Srow[w][0][0];  // stride 136 f16
    v16h ap0 = load_frag(pbase, 136);
    v16h ap1 = load_frag(pbase + 32, 136);
#pragma unroll
    for (int ct = 0; ct < 4; ++ct) {
      v8f acc = zero8();
      acc = wmma_f16(ap0, load_frag(&vp[ct * 16][0], 66), acc);
      acc = wmma_f16(ap1, load_frag(&vp[ct * 16][32], 66), acc);
#pragma unroll
      for (int i = 0; i < 8; ++i) {
        const int n  = m0 + i + (hf << 3);
        const int hd = ct * 16 + r;
        const int f  = ((hd * NH + h) << 12) + n;  // faithful permute+reshape
        const int n2 = f >> 8, c2 = f & 255;
        xsa[((size_t)(b << 12) + n2) * NC + c2] = (_Float16)acc[i];
      }
    }
    __syncthreads();
  }
}

// ---------------------------------------------------------------------------
// Host launcher. Workspace (~304 MB of f16 intermediates) carved from d_ws.
// ---------------------------------------------------------------------------
extern "C" void kernel_launch(void* const* d_in, const int* in_sizes, int n_in,
                              void* d_out, int out_size, void* d_ws,
                              size_t ws_size, hipStream_t stream) {
  const float* x     = (const float*)d_in[0];
  const float* ref   = (const float*)d_in[1];
  const float* Wq    = (const float*)d_in[2];
  const float* bq    = (const float*)d_in[3];
  const float* Wkvv  = (const float*)d_in[4];
  const float* bkvv  = (const float*)d_in[5];
  const float* WE    = (const float*)d_in[6];
  const float* bE    = (const float*)d_in[7];
  const float* Wo1   = (const float*)d_in[8];
  const float* bo1   = (const float*)d_in[9];
  const float* Wo2   = (const float*)d_in[10];
  const float* bo2   = (const float*)d_in[11];
  const float* temp  = (const float*)d_in[12];
  const float* temp2 = (const float*)d_in[13];
  float* out = (float*)d_out;

  const size_t TSE = (size_t)NB * NH * HD * NTOK;  // 16,777,216 elems
  const size_t TSB = TSE * sizeof(_Float16);       // 32 MiB per tensor
  char* ws = (char*)d_ws;
  _Float16* qraw   = (_Float16*)ws;  ws += TSB;
  _Float16* kvv    = (_Float16*)ws;  ws += 3 * TSB;  // k | v_ca(tok) | v_sa
  _Float16* qn     = (_Float16*)ws;  ws += TSB;
  _Float16* kn     = (_Float16*)ws;  ws += TSB;
  _Float16* qnt    = (_Float16*)ws;  ws += TSB;
  _Float16* xca    = (_Float16*)ws;  ws += TSB;
  _Float16* xsa    = (_Float16*)ws;  ws += TSB;
  _Float16* WEh    = (_Float16*)ws;  ws += (size_t)NP * NTOK * 2;
  _Float16* kprojT = (_Float16*)ws;  ws += (size_t)NB * NH * NP * HD * 2;
  _Float16* vproj  = (_Float16*)ws;  ws += (size_t)NB * NH * NP * HD * 2;

  // 1) WE -> f16
  cvt_f32_f16_kernel<<<(NP * NTOK + 255) / 256, 256, 0, stream>>>(
      WE, WEh, NP * NTOK);
  // 2) q = x @ Wq^T + bq   (M=65536, Nc=256, K=256)
  gemm_kernel<1, false><<<dim3(512, 2), 256, 0, stream>>>(
      x, Wq, bq, NC, nullptr, qraw);
  // 3) kvv = ref @ Wkvv^T + bkvv  (Nc=768)
  gemm_kernel<2, false><<<dim3(512, 6), 256, 0, stream>>>(
      ref, Wkvv, bkvv, NC, nullptr, kvv);
  // 4) token-axis L2 normalize q,k
  norm_kernel<<<dim3(NB * NH * HD, 2), 256, 0, stream>>>(
      qraw, kvv /*k*/, qn, kn, qnt);
  // 5) channel attention
  ca_attn_kernel<<<NB * NH, 256, 0, stream>>>(
      qn, kn, kvv + TSE /*v_ca*/, temp, xca);
  // 6) Linformer projections
  linproj_kernel<<<NB * NH, 256, 0, stream>>>(
      kn, kvv + 2 * TSE /*v_sa*/, WEh, bE, kprojT, vproj);
  // 7) spatial attention
  sa_attn_kernel<<<NB * NH, 256, 0, stream>>>(qnt, kprojT, vproj, temp2, xsa);
  // 8) output projections: cols [0,128)=SA (Wo2), [128,256)=CA (Wo1)
  //    (A operands are f16 -> staged to LDS via the Tensor Data Mover)
  gemm_kernel<0, true><<<dim3(512, 1), 256, 0, stream>>>(
      xsa, Wo2, bo2, NC, out, nullptr);
  gemm_kernel<0, true><<<dim3(512, 1), 256, 0, stream>>>(
      xca, Wo1, bo1, NC, out + 128, nullptr);
}
